// MFITSRec_36859409334694
// MI455X (gfx1250) — matrix operand order repaired and checked by reading, not compiled, not run on hardware
//
#include <hip/hip_runtime.h>
#include <cstdint>
#include <cstddef>

// ---------------------------------------------------------------------------
// MFITSRec (TiSASRec-like) forward for MI455X (gfx1250, wave32, WMMA).
// All GEMM-shaped math runs through v_wmma_f32_16x16x32_bf16 (f32 accumulate).
// WMMA fragments are fed by 16-byte vector loads; fragment loads are
// software-pipelined ahead of the wmma chains; gathers use int4/float4.
// ---------------------------------------------------------------------------

typedef __bf16 bf16_t;
typedef __attribute__((ext_vector_type(16))) __bf16 v16bf;
typedef __attribute__((ext_vector_type(8)))  __bf16 v8bf;
typedef __attribute__((ext_vector_type(8)))  float  v8f;

#define NBATCH 16
#define LSEQ   200
#define HID    128
#define NHEADS 4
#define DHEAD  32
#define LPAD   224              // 14 tiles of 16; multiple of 32 for WMMA K-chunks
#define NROWS  (NBATCH * LSEQ)  // 3200

__device__ __forceinline__ float wave_sum32(float v) {
#pragma unroll
  for (int off = 16; off > 0; off >>= 1) v += __shfl_xor(v, off, 32);
  return v;
}

// 16 contiguous bf16 -> v16bf (two b128 loads)
__device__ __forceinline__ v16bf load16(const bf16_t* p) {
  v8bf lo = *(const v8bf*)p;
  v8bf hi = *(const v8bf*)(p + 8);
  return __builtin_shufflevector(lo, hi, 0, 1, 2, 3, 4, 5, 6, 7,
                                 8, 9, 10, 11, 12, 13, 14, 15);
}
// two separate 8-half contiguous chunks -> v16bf
__device__ __forceinline__ v16bf load8x2(const bf16_t* p0, const bf16_t* p1) {
  v8bf lo = *(const v8bf*)p0;
  v8bf hi = *(const v8bf*)p1;
  return __builtin_shufflevector(lo, hi, 0, 1, 2, 3, 4, 5, 6, 7,
                                 8, 9, 10, 11, 12, 13, 14, 15);
}

// ---------------------------------------------------------------------------
// 1) seqs[b,l,:] = concat(item_emb, b_emb, c_emb)[log_seqs] * sqrt(96) * valid
// ---------------------------------------------------------------------------
__global__ __launch_bounds__(128) void k_embed_seqs(
    const int* __restrict__ log_seqs, const int* __restrict__ item_meta,
    const float* __restrict__ ie, const float* __restrict__ ibw,
    const float* __restrict__ icw, float* __restrict__ seqs,
    bf16_t* __restrict__ seqs_bf) {
  int row = blockIdx.x;           // b*L + l
  int d   = threadIdx.x;          // 0..127
  int id  = log_seqs[row];
  float v;
  if (d < 96)       v = ie[(size_t)id * 96 + d];
  else if (d < 112) v = ibw[(size_t)item_meta[id * 2 + 0] * 16 + (d - 96)];
  else              v = icw[(size_t)item_meta[id * 2 + 1] * 16 + (d - 112)];
  v *= 9.797958971132712f;        // sqrt(96)
  if (id == 0) v = 0.f;
  seqs[(size_t)row * HID + d]    = v;
  seqs_bf[(size_t)row * HID + d] = (bf16_t)v;
}

// ---------------------------------------------------------------------------
// 2) LayerNorm over last dim (128). One wave32 per row; lane holds 4 elems.
// ---------------------------------------------------------------------------
__global__ __launch_bounds__(256) void k_layernorm(
    const float* __restrict__ x, const float* __restrict__ g,
    const float* __restrict__ be, float* __restrict__ yF,
    bf16_t* __restrict__ yB, int nrows) {
  int wave = threadIdx.x >> 5, lane = threadIdx.x & 31;
  int row = blockIdx.x * 8 + wave;
  if (row >= nrows) return;
  const float* xr = x + (size_t)row * HID;
  float v[4];
  float s = 0.f;
#pragma unroll
  for (int k = 0; k < 4; ++k) { v[k] = xr[lane + 32 * k]; s += v[k]; }
  float mu = wave_sum32(s) * (1.f / 128.f);
  float var = 0.f;
#pragma unroll
  for (int k = 0; k < 4; ++k) { float d = v[k] - mu; var += d * d; }
  var = wave_sum32(var) * (1.f / 128.f);
  float rstd = rsqrtf(var + 1e-8f);
#pragma unroll
  for (int k = 0; k < 4; ++k) {
    int d = lane + 32 * k;
    float y = (v[k] - mu) * rstd * g[d] + be[d];
    if (yF) yF[(size_t)row * HID + d] = y;
    if (yB) yB[(size_t)row * HID + d] = (bf16_t)y;
  }
}

// ---------------------------------------------------------------------------
// 3) C[M,128] = act(A_bf16[M,128] @ W_f32[128,128] + bias) (+residual)(*mask)
//    Epilogue behavior is a compile-time template -> branch-free stores.
//    All 4 A-fragments and 4 B-fragments preloaded, then 4 chained wmmas.
// ---------------------------------------------------------------------------
template <int RELU, int HASRES, int HASMASK, int OUTF, int OUTB>
__global__ __launch_bounds__(256) void k_gemm128(
    const bf16_t* __restrict__ A, const float* __restrict__ W,
    const float* __restrict__ bias, const float* __restrict__ residual,
    const int* __restrict__ maskids,
    float* __restrict__ outF, bf16_t* __restrict__ outB, int M) {
  __shared__ __align__(16) bf16_t sA[64 * 128];    // 16 KB
  __shared__ __align__(16) bf16_t sWT[128 * 128];  // 32 KB, [n][k]
  int tid = threadIdx.x;
  int rowBase = blockIdx.x * 64;

  for (int i = tid; i < 128 * 128; i += 256) {
    int k = i >> 7, n = i & 127;
    sWT[n * 128 + k] = (bf16_t)W[i];               // coalesced read, transpose
  }
  for (int i = tid; i < 64 * 128; i += 256) {
    int r = i >> 7, c = i & 127;
    int gr = rowBase + r;
    sA[i] = (gr < M) ? A[(size_t)gr * 128 + c] : (bf16_t)0.f;
  }
  if (rowBase + 64 < M) __builtin_prefetch(A + (size_t)(rowBase + 64) * 128);
  __syncthreads();

  int wave = tid >> 5, lane = tid & 31;
  int mt = wave & 3;                   // 4 m-tiles of 16 rows
  int ntBase = (wave >> 2) * 4;        // each half of waves takes 4 n-tiles
  int arow = mt * 16 + (lane & 15);

  v16bf afrs[4];
#pragma unroll
  for (int kk = 0; kk < 4; ++kk) {
    int ac = kk * 32 + ((lane & 16) ? 16 : 0);
    afrs[kk] = load16(&sA[arow * 128 + ac]);
  }

  for (int nt = ntBase; nt < ntBase + 4; ++nt) {
    int ncol = nt * 16 + (lane & 15);
    v16bf bfrs[4];
#pragma unroll
    for (int kk = 0; kk < 4; ++kk)
      bfrs[kk] = load16(&sWT[ncol * 128 + kk * 32 + ((lane & 16) ? 16 : 0)]);
    v8f acc = {};
#pragma unroll
    for (int kk = 0; kk < 4; ++kk)
      acc = __builtin_amdgcn_wmma_f32_16x16x32_bf16(false, afrs[kk], false,
                                                    bfrs[kk], (short)0, acc,
                                                    false, false);
    float bv = bias[ncol];
#pragma unroll
    for (int r = 0; r < 8; ++r) {
      int rr = mt * 16 + ((lane & 16) ? r + 8 : r);
      int grow = rowBase + rr;
      float v = acc[r] + bv;
      if (RELU) v = fmaxf(v, 0.f);
      if (HASRES) v += residual[(size_t)grow * 128 + ncol];
      if (HASMASK) { if (maskids[grow] == 0) v = 0.f; }
      if (OUTF) outF[(size_t)grow * 128 + ncol] = v;
      if (OUTB) outB[(size_t)grow * 128 + ncol] = (bf16_t)v;
    }
  }
}

// ---------------------------------------------------------------------------
// 4) Repack q/k into head-major padded bf16 [B][NH][LPAD][32]; (V+posV) is
//    stored TRANSPOSED [B][NH][32][LPAD] so attention-output B fragments are
//    contiguous 16-byte loads.
// ---------------------------------------------------------------------------
__global__ __launch_bounds__(256) void k_rearrange(
    const float* __restrict__ q, const float* __restrict__ k,
    const float* __restrict__ v, const float* __restrict__ posV,
    bf16_t* __restrict__ q_bf, bf16_t* __restrict__ k_bf,
    bf16_t* __restrict__ vpvT_bf) {
  int idx = blockIdx.x * blockDim.x + threadIdx.x;
  const int total = NBATCH * NHEADS * LPAD * 32;
  if (idx >= total) return;
  int d = idx & 31;
  int j = (idx >> 5) % LPAD;
  int h = ((idx >> 5) / LPAD) % NHEADS;
  int b = idx / (NHEADS * LPAD * 32);
  size_t tidx = (((size_t)(b * NHEADS + h)) * 32 + d) * LPAD + j;
  if (j < LSEQ) {
    size_t src = ((size_t)(b * LSEQ + j)) * HID + h * DHEAD + d;
    q_bf[idx]     = (bf16_t)q[src];
    k_bf[idx]     = (bf16_t)k[src];
    vpvT_bf[tidx] = (bf16_t)(v[src] + posV[(size_t)j * HID + h * DHEAD + d]);
  } else {
    q_bf[idx]     = (bf16_t)0.f;
    k_bf[idx]     = (bf16_t)0.f;
    vpvT_bf[tidx] = (bf16_t)0.f;
  }
}

__global__ __launch_bounds__(256) void k_posk(
    const float* __restrict__ posK, bf16_t* __restrict__ posK_bf) {
  int idx = blockIdx.x * blockDim.x + threadIdx.x;
  const int total = NHEADS * LPAD * 32;
  if (idx >= total) return;
  int d = idx & 31;
  int j = (idx >> 5) % LPAD;
  int h = (idx >> 5) / LPAD;
  posK_bf[idx] = (j < LSEQ) ? (bf16_t)posK[(size_t)j * HID + h * DHEAD + d]
                            : (bf16_t)0.f;
}

// ---------------------------------------------------------------------------
// 5) Attention, one workgroup (4 waves) per (b,h).
//    Scores: WMMA(Q,K^T) + WMMA(Q,posK^T) + gathered q.timeK[t] ; softmax
//    Output: WMMA(A, V+posV) + gathered A.timeV[t]
// ---------------------------------------------------------------------------
__global__ __launch_bounds__(128) void k_attention(
    const bf16_t* __restrict__ q_bf, const bf16_t* __restrict__ k_bf,
    const bf16_t* __restrict__ vpvT_bf, const bf16_t* __restrict__ posK_bf,
    const float* __restrict__ time_K_W, const float* __restrict__ time_V_W,
    const int* __restrict__ time_matrices, const float* __restrict__ qf,
    float* __restrict__ attn) {
  __shared__ __align__(16) float S[4][16][LPAD];   // 56 KB

  int bh = blockIdx.x;
  int b = bh >> 2, h = bh & 3;
  int wave = threadIdx.x >> 5, lane = threadIdx.x & 31;

  const bf16_t* Qh  = q_bf    + ((size_t)(b * NHEADS + h)) * LPAD * 32;
  const bf16_t* Kh  = k_bf    + ((size_t)(b * NHEADS + h)) * LPAD * 32;
  const bf16_t* VhT = vpvT_bf + ((size_t)(b * NHEADS + h)) * 32 * LPAD;
  const bf16_t* Ph  = posK_bf + (size_t)h * LPAD * 32;
  const int* TM = time_matrices + (size_t)b * LSEQ * LSEQ;
  const float* tKh = time_K_W + h * DHEAD;
  const float* tVh = time_V_W + h * DHEAD;

  for (int qt0 = 0; qt0 < 4; ++qt0) {         // 13 q-tiles over 4 waves
    int qt = qt0 * 4 + wave;
    bool act = (qt < 13);
    int i0 = qt * 16;

    if (act) {
      // A-fragment of Q (16 rows x K=32): 16 contiguous k per lane
      const bf16_t* qr = Qh + (size_t)(i0 + (lane & 15)) * 32;
      int c = (lane & 16) ? 16 : 0;
      v16bf qa = load16(qr + c);
      int kb = (lane & 16) ? 16 : 0;
      // Scores: S = Q@K^T + Q@posK^T ; fragment loads pipelined 1 tile ahead
      v16bf kfr = load16(Kh + (size_t)(lane & 15) * 32 + kb);
      v16bf pfr = load16(Ph + (size_t)(lane & 15) * 32 + kb);
      for (int jt = 0; jt < 14; ++jt) {
        v16bf kcur = kfr, pcur = pfr;
        if (jt < 13) {
          int jrow = (jt + 1) * 16 + (lane & 15);
          kfr = load16(Kh + (size_t)jrow * 32 + kb);
          pfr = load16(Ph + (size_t)jrow * 32 + kb);
        }
        v8f acc = {};
        acc = __builtin_amdgcn_wmma_f32_16x16x32_bf16(false, qa, false, kcur,
                                                      (short)0, acc, false, false);
        acc = __builtin_amdgcn_wmma_f32_16x16x32_bf16(false, qa, false, pcur,
                                                      (short)0, acc, false, false);
#pragma unroll
        for (int r = 0; r < 8; ++r)
          S[wave][(lane & 16) ? r + 8 : r][jt * 16 + (lane & 15)] = acc[r];
      }
    }
    __syncthreads();

    if (act) {
      // Scalar phase: relative-time score gather (int4 indices, float4 rows),
      // mask, scale, softmax.  Two lanes per row, 112 cols each.
      int rowi = lane & 15;
      int i = i0 + rowi;
      bool irow_ok = (i < LSEQ);
      int j0 = (lane >> 4) * 112;
      float* Srow = &S[wave][rowi][0];
      float qv[DHEAD];
      if (irow_ok) {
        const float* qrow = qf + ((size_t)(b * LSEQ + i)) * HID + h * DHEAD;
#pragma unroll
        for (int d = 0; d < DHEAD; d += 4) {
          float4 f = *(const float4*)(qrow + d);
          qv[d] = f.x; qv[d + 1] = f.y; qv[d + 2] = f.z; qv[d + 3] = f.w;
        }
      }
      const int* tmrow = TM + (size_t)(irow_ok ? i : 0) * LSEQ;
      for (int jb = j0; jb < j0 + 112; jb += 4) {
        int4 t4 = make_int4(0, 0, 0, 0);
        if (irow_ok && jb < LSEQ) t4 = *(const int4*)(tmrow + jb);
        int ts[4] = {t4.x, t4.y, t4.z, t4.w};
#pragma unroll
        for (int u = 0; u < 4; ++u) {
          int j = jb + u;
          float sv = Srow[j];
          if (j >= LSEQ || j > i) {
            sv = -1e30f;               // causal + padding mask
          } else {
            if (irow_ok) {
              const float* tk = tKh + (size_t)ts[u] * HID;
              float ds = 0.f;
#pragma unroll
              for (int d = 0; d < DHEAD; d += 4) {
                float4 t = *(const float4*)(tk + d);
                ds += qv[d] * t.x + qv[d + 1] * t.y + qv[d + 2] * t.z +
                      qv[d + 3] * t.w;
              }
              sv += ds;
            }
            sv *= 0.17677669529663687f;  // 1/sqrt(32)
          }
          Srow[j] = sv;
        }
      }
      float m = -1e30f;
      for (int j = j0; j < j0 + 112; ++j) m = fmaxf(m, Srow[j]);
      m = fmaxf(m, __shfl_xor(m, 16, 32));
      float sum = 0.f;
      for (int j = j0; j < j0 + 112; ++j) {
        float e = __expf(Srow[j] - m);
        Srow[j] = e;
        sum += e;
      }
      sum += __shfl_xor(sum, 16, 32);
      float inv = 1.f / sum;
      for (int j = j0; j < j0 + 112; ++j) Srow[j] *= inv;
    }
    __syncthreads();

    if (act) {
      // Output: O = A @ (V+posV); V fragment loads pipelined 1 chunk ahead.
      int rowi = lane & 15;
#pragma unroll
      for (int nt = 0; nt < 2; ++nt) {
        int ncol = nt * 16 + (lane & 15);
        int kboff = (lane & 16) ? 16 : 0;
        v16bf bfr = load16(&VhT[(size_t)ncol * LPAD + kboff]);
        v8f oacc = {};
        for (int kc = 0; kc < 7; ++kc) {
          v16bf bcur = bfr;
          if (kc < 6)
            bfr = load16(&VhT[(size_t)ncol * LPAD + (kc + 1) * 32 + kboff]);
          int c = kc * 32 + ((lane & 16) ? 16 : 0);
          v16bf afr;
#pragma unroll
          for (int p = 0; p < 8; ++p) {
            afr[p]     = (bf16_t)S[wave][rowi][c + p];
            afr[p + 8] = (bf16_t)S[wave][rowi][c + 8 + p];
          }
          oacc = __builtin_amdgcn_wmma_f32_16x16x32_bf16(false, afr, false, bcur,
                                                         (short)0, oacc, false, false);
        }
#pragma unroll
        for (int r = 0; r < 8; ++r) {
          int rr = i0 + ((lane & 16) ? r + 8 : r);
          if (rr < LSEQ)
            attn[((size_t)(b * LSEQ + rr)) * HID + h * DHEAD + ncol] = oacc[r];
        }
      }
      // Relative-time output gather: out_i += sum_j A_ij * timeV[t_bij]
      int i = i0 + (lane & 15);
      if (i < LSEQ) {
        int dbase = (lane >> 4) * 16;
        float* orow = attn + ((size_t)(b * LSEQ + i)) * HID + h * DHEAD + dbase;
        float acc[16];
#pragma unroll
        for (int d = 0; d < 16; ++d) acc[d] = orow[d];
        const int* tmrow = TM + (size_t)i * LSEQ;
        const float* Srow = &S[wave][lane & 15][0];
        for (int jb = 0; jb <= i; jb += 4) {
          int4 t4 = *(const int4*)(tmrow + jb);
          int ts[4] = {t4.x, t4.y, t4.z, t4.w};
#pragma unroll
          for (int u = 0; u < 4; ++u) {
            int j = jb + u;
            if (j <= i) {
              float a = Srow[j];
              const float* tv = tVh + (size_t)ts[u] * HID + dbase;
#pragma unroll
              for (int d = 0; d < 16; d += 4) {
                float4 t = *(const float4*)(tv + d);
                acc[d]     += a * t.x;
                acc[d + 1] += a * t.y;
                acc[d + 2] += a * t.z;
                acc[d + 3] += a * t.w;
              }
            }
          }
        }
#pragma unroll
        for (int d = 0; d < 16; ++d) orow[d] = acc[d];
      }
    }
    __syncthreads();
  }
}

// ---------------------------------------------------------------------------
// 6) Elementwise residual add
// ---------------------------------------------------------------------------
__global__ __launch_bounds__(256) void k_add(const float* __restrict__ a,
                                             const float* __restrict__ b,
                                             float* __restrict__ out, int n) {
  int i = blockIdx.x * blockDim.x + threadIdx.x;
  if (i < n) out[i] = a[i] + b[i];
}

// ---------------------------------------------------------------------------
// 7) Final logits: dot(log_feats, embed(pos/neg)). One wave per (row, pos/neg).
// ---------------------------------------------------------------------------
__global__ __launch_bounds__(64) void k_logits(
    const float* __restrict__ lf, const int* __restrict__ pos,
    const int* __restrict__ neg, const int* __restrict__ meta,
    const float* __restrict__ ie, const float* __restrict__ ibw,
    const float* __restrict__ icw, float* __restrict__ out) {
  int row = blockIdx.x;
  int wave = threadIdx.x >> 5, lane = threadIdx.x & 31;
  int id = (wave ? neg : pos)[row];
  int bb = meta[id * 2 + 0], cc = meta[id * 2 + 1];
  const float* lr = lf + (size_t)row * HID;
  float s = 0.f;
#pragma unroll
  for (int k = 0; k < 4; ++k) {
    int d = lane + 32 * k;
    float e;
    if (d < 96)       e = ie[(size_t)id * 96 + d];
    else if (d < 112) e = ibw[(size_t)bb * 16 + (d - 96)];
    else              e = icw[(size_t)cc * 16 + (d - 112)];
    s += e * lr[d];
  }
  s = wave_sum32(s);
  if (lane == 0) out[wave * NROWS + row] = s;
}

// ---------------------------------------------------------------------------
// Host-side orchestration
// ---------------------------------------------------------------------------
extern "C" void kernel_launch(void* const* d_in, const int* in_sizes, int n_in,
                              void* d_out, int out_size, void* d_ws, size_t ws_size,
                              hipStream_t stream) {
  (void)in_sizes; (void)n_in; (void)out_size; (void)ws_size;
  const int*   log_seqs   = (const int*)d_in[1];
  const int*   time_mat   = (const int*)d_in[2];
  const int*   pos_seqs   = (const int*)d_in[3];
  const int*   neg_seqs   = (const int*)d_in[4];
  const int*   item_meta  = (const int*)d_in[5];
  const float* item_emb_W = (const float*)d_in[6];
  const float* item_b_W   = (const float*)d_in[7];
  const float* item_c_W   = (const float*)d_in[8];
  const float* abs_pos_K  = (const float*)d_in[9];
  const float* abs_pos_V  = (const float*)d_in[10];
  const float* time_K_W   = (const float*)d_in[11];
  const float* time_V_W   = (const float*)d_in[12];
  const float* attn_ln_s  = (const float*)d_in[13];
  const float* attn_ln_b  = (const float*)d_in[14];
  const float* Qw = (const float*)d_in[15];
  const float* Qb = (const float*)d_in[16];
  const float* Kw = (const float*)d_in[17];
  const float* Kb = (const float*)d_in[18];
  const float* Vw = (const float*)d_in[19];
  const float* Vb = (const float*)d_in[20];
  const float* fwd_ln_s = (const float*)d_in[21];
  const float* fwd_ln_b = (const float*)d_in[22];
  const float* c1w = (const float*)d_in[23];
  const float* c1b = (const float*)d_in[24];
  const float* c2w = (const float*)d_in[25];
  const float* c2b = (const float*)d_in[26];
  const float* last_ln_s = (const float*)d_in[27];
  const float* last_ln_b = (const float*)d_in[28];
  float* out = (float*)d_out;

  // ---- workspace carve (bytes, 256-aligned) ----
  char* w = (char*)d_ws;
  auto alloc = [&](size_t bytes) -> char* {
    char* p = w;
    w += (bytes + 255) & ~(size_t)255;
    return p;
  };
  const size_t F = (size_t)NROWS * HID;     // 409600 elements
  float*  seqs    = (float*)alloc(F * 4);
  float*  Qn      = (float*)alloc(F * 4);
  float*  qbuf    = (float*)alloc(F * 4);
  float*  kbuf    = (float*)alloc(F * 4);
  float*  vbuf    = (float*)alloc(F * 4);
  float*  attn    = (float*)alloc(F * 4);
  float*  xbuf    = (float*)alloc(F * 4);
  bf16_t* seqs_bf = (bf16_t*)alloc(F * 2);
  bf16_t* Qn_bf   = (bf16_t*)alloc(F * 2);
  bf16_t* x_bf    = (bf16_t*)alloc(F * 2);
  bf16_t* h1_bf   = (bf16_t*)alloc(F * 2);
  const size_t H = (size_t)NBATCH * NHEADS * LPAD * 32;   // 458752
  bf16_t* qh_bf    = (bf16_t*)alloc(H * 2);
  bf16_t* kh_bf    = (bf16_t*)alloc(H * 2);
  bf16_t* vpvT_bf  = (bf16_t*)alloc(H * 2);
  bf16_t* posk_bf  = (bf16_t*)alloc((size_t)NHEADS * LPAD * 32 * 2);

  // ---- pipeline ----
  k_embed_seqs<<<NROWS, 128, 0, stream>>>(log_seqs, item_meta, item_emb_W,
                                          item_b_W, item_c_W, seqs, seqs_bf);
  k_posk<<<(NHEADS * LPAD * 32) / 256, 256, 0, stream>>>(abs_pos_K, posk_bf);

  for (int blk = 0; blk < 2; ++blk) {
    const float* als = attn_ln_s + blk * HID;
    const float* alb = attn_ln_b + blk * HID;
    const float* qw = Qw + (size_t)blk * HID * HID;
    const float* kw = Kw + (size_t)blk * HID * HID;
    const float* vw = Vw + (size_t)blk * HID * HID;
    const float* qb = Qb + blk * HID;
    const float* kb = Kb + blk * HID;
    const float* vb = Vb + blk * HID;

    // Qn = LN(seqs)
    k_layernorm<<<NROWS / 8, 256, 0, stream>>>(seqs, als, alb, Qn, Qn_bf, NROWS);
    // q/k/v projections (WMMA, plain epilogue)
    k_gemm128<0, 0, 0, 1, 0><<<NROWS / 64, 256, 0, stream>>>(
        Qn_bf, qw, qb, nullptr, nullptr, qbuf, nullptr, NROWS);
    k_gemm128<0, 0, 0, 1, 0><<<NROWS / 64, 256, 0, stream>>>(
        seqs_bf, kw, kb, nullptr, nullptr, kbuf, nullptr, NROWS);
    k_gemm128<0, 0, 0, 1, 0><<<NROWS / 64, 256, 0, stream>>>(
        seqs_bf, vw, vb, nullptr, nullptr, vbuf, nullptr, NROWS);
    // head-major bf16 repack (+posV fold, V transposed)
    k_rearrange<<<(int)(H / 256), 256, 0, stream>>>(qbuf, kbuf, vbuf, abs_pos_V,
                                                    qh_bf, kh_bf, vpvT_bf);
    // attention
    k_attention<<<NBATCH * NHEADS, 128, 0, stream>>>(
        qh_bf, kh_bf, vpvT_bf, posk_bf, time_K_W, time_V_W, time_mat, qbuf, attn);
    // seqs = Qn + attn_out
    k_add<<<(int)(F / 256), 256, 0, stream>>>(Qn, attn, seqs, (int)F);
    // FFN: x = LN(seqs); seqs = (x + relu(x@c1+b1)@c2 + b2) * valid
    k_layernorm<<<NROWS / 8, 256, 0, stream>>>(seqs, fwd_ln_s + blk * HID,
                                               fwd_ln_b + blk * HID, xbuf, x_bf,
                                               NROWS);
    k_gemm128<1, 0, 0, 0, 1><<<NROWS / 64, 256, 0, stream>>>(
        x_bf, c1w + (size_t)blk * HID * HID, c1b + blk * HID, nullptr, nullptr,
        nullptr, h1_bf, NROWS);
    k_gemm128<0, 1, 1, 1, 1><<<NROWS / 64, 256, 0, stream>>>(
        h1_bf, c2w + (size_t)blk * HID * HID, c2b + blk * HID, xbuf, log_seqs,
        seqs, seqs_bf, NROWS);
  }

  // log_feats = LN(seqs, last); logits
  k_layernorm<<<NROWS / 8, 256, 0, stream>>>(seqs, last_ln_s, last_ln_b, xbuf,
                                             nullptr, NROWS);
  k_logits<<<NROWS, 64, 0, stream>>>(xbuf, pos_seqs, neg_seqs, item_meta,
                                     item_emb_W, item_b_W, item_c_W, out);
}
